// SimpleTransformer_88656714924897
// MI455X (gfx1250) — matrix operand order repaired and checked by reading, not compiled, not run on hardware
//
#include <hip/hip_runtime.h>
#include <hip/hip_fp16.h>

// Problem constants (match reference)
#define BB   2
#define SS   1024
#define EE   1024
#define HH   8
#define HIDD 2048
#define VV   32000
#define DKK  128
#define MM   (BB*SS)   // 2048 rows

typedef __attribute__((ext_vector_type(16))) _Float16 v16h;
typedef __attribute__((ext_vector_type(8)))  _Float16 v8h;
typedef __attribute__((ext_vector_type(8)))  float    v8f;

#define WMMA_F16(a,b,c) \
  __builtin_amdgcn_wmma_f32_16x16x32_f16(false,(a),false,(b),(short)0,(c),false,false)

// ---------------------------------------------------------------------------
// Embedding gather + positional encoding:  out[r] = emb[tok[r]] + pos[r % S]
// ---------------------------------------------------------------------------
__global__ __launch_bounds__(256)
void embed_kernel(const int* __restrict__ tok, const float* __restrict__ emb,
                  const float* __restrict__ pos, float* __restrict__ out)
{
    const int rs = blockIdx.x;           // b*S + s
    const int s  = rs & (SS - 1);
    const int t  = tok[rs];
    const float* e = emb + (size_t)t * EE;
    const float* p = pos + (size_t)s * EE;
    float* o = out + (size_t)rs * EE;
    for (int i = threadIdx.x; i < EE; i += 256) o[i] = e[i] + p[i];
}

// ---------------------------------------------------------------------------
// WMMA GEMM: C[M,N] = A[M,K] * B[K,N] + bias[N] (+ res[M,N]) (+ ReLU)
// fp32 in/out, f16 WMMA compute with fp32 accumulate.
// Block tile 128x128, K-step 32, 8 waves -> each wave 32(M) x 64(N).
// HAS_RES / RELU are compile-time so the epilogue is branch-free.
// ---------------------------------------------------------------------------
#define BM 128
#define BN 128
#define BK 32
#define LDP 40   // padded LDS row stride in halfs (80B, 16B aligned)

template <bool HAS_RES, bool RELU>
__global__ __launch_bounds__(256)
void gemm_wmma_kernel(const float* __restrict__ A, const float* __restrict__ Bm,
                      const float* __restrict__ bias, const float* __restrict__ res,
                      float* __restrict__ C, int M, int N, int K)
{
    __shared__ _Float16 sA [BM][LDP];   // A tile, [m][k]
    __shared__ _Float16 sBT[BN][LDP];   // B tile transposed, [n][k]

    const int tid  = threadIdx.x;
    const int wave = tid >> 5;
    const int lane = tid & 31;
    const int l16  = lane & 15;
    const int h16  = lane >> 4;         // 0 | 1

    const int bm = blockIdx.y * BM;
    const int bn = blockIdx.x * BN;
    const int wm = (wave & 3) * 32;     // wave M offset within block
    const int wn = (wave >> 2) * 64;    // wave N offset within block

    // B staging ownership: one column per thread, two k-quads in parallel
    const int sn = tid & 127;           // column within block tile
    const int kg = tid >> 7;            // 0 | 1  (k-quad phase)

    v8f acc[2][4];
#pragma unroll
    for (int i = 0; i < 2; ++i)
#pragma unroll
        for (int j = 0; j < 4; ++j) acc[i][j] = (v8f)(0.0f);

    for (int k0 = 0; k0 < K; k0 += BK) {
        // ---- phase 1: issue ALL global loads for this tile (stay in flight)
        float4 fa[4];
#pragma unroll
        for (int i = 0; i < 4; ++i) {
            const int idx = tid + i * 256;
            const int r = idx >> 3;           // 8 float4 per 32-wide row
            const int c = (idx & 7) * 4;
            fa[i] = *(const float4*)(A + (size_t)(bm + r) * K + k0 + c);
        }
        float fb[4][4];                       // B columns: 4 quads of 4 k-values
#pragma unroll
        for (int q = 0; q < 4; ++q) {
            const int kq = q * 8 + kg * 4;
#pragma unroll
            for (int j = 0; j < 4; ++j)
                fb[q][j] = Bm[(size_t)(k0 + kq + j) * N + bn + sn];
        }
        if (k0 + BK < K) {  // CDNA5 speculative prefetch of the next K tile
            __builtin_prefetch(A  + (size_t)(bm + (tid & 127)) * K + k0 + BK, 0, 1);
            __builtin_prefetch(Bm + (size_t)(k0 + BK + (tid & 31)) * N + bn, 0, 1);
        }
        // ---- phase 2: convert + packed LDS stores (b64-mergeable)
#pragma unroll
        for (int i = 0; i < 4; ++i) {
            const int idx = tid + i * 256;
            const int r = idx >> 3;
            const int c = (idx & 7) * 4;
            sA[r][c + 0] = (_Float16)fa[i].x; sA[r][c + 1] = (_Float16)fa[i].y;
            sA[r][c + 2] = (_Float16)fa[i].z; sA[r][c + 3] = (_Float16)fa[i].w;
        }
#pragma unroll
        for (int q = 0; q < 4; ++q) {
            const int kq = q * 8 + kg * 4;
            _Float16* d = &sBT[sn][kq];       // 8B-aligned, contiguous 4 halfs
            d[0] = (_Float16)fb[q][0]; d[1] = (_Float16)fb[q][1];
            d[2] = (_Float16)fb[q][2]; d[3] = (_Float16)fb[q][3];
        }
        __syncthreads();

        // ---- A fragments (16-bit A layout: VGPR0-3 -> K=kb8+0..7, VGPR4-7 -> 16+kb8+0..7)
        v16h afr[2];
#pragma unroll
        for (int mt = 0; mt < 2; ++mt) {
            const _Float16* pa = &sA[wm + mt * 16 + l16][h16 * 8];
            const v8h lo = *(const v8h*)pa;
            const v8h hi = *(const v8h*)(pa + 16);
            v16h a;
#pragma unroll
            for (int i = 0; i < 8; ++i) { a[i] = lo[i]; a[i + 8] = hi[i]; }
            afr[mt] = a;
        }
        // ---- B fragments (same K order as A: kb8+0..7, 16+kb8+0..7; col = l16)
        v16h bfr[4];
#pragma unroll
        for (int nt = 0; nt < 4; ++nt) {
            const _Float16* pb = &sBT[wn + nt * 16 + l16][h16 * 8];
            const v8h lo = *(const v8h*)pb;
            const v8h hi = *(const v8h*)(pb + 16);
            v16h bvx;
#pragma unroll
            for (int i = 0; i < 8; ++i) { bvx[i] = lo[i]; bvx[i + 8] = hi[i]; }
            bfr[nt] = bvx;
        }
#pragma unroll
        for (int mt = 0; mt < 2; ++mt)
#pragma unroll
            for (int nt = 0; nt < 4; ++nt)
                acc[mt][nt] = WMMA_F16(afr[mt], bfr[nt], acc[mt][nt]);
        __syncthreads();
    }

    // ---- epilogue (branch-free): bias (+res)(+relu)
#pragma unroll
    for (int mt = 0; mt < 2; ++mt)
#pragma unroll
        for (int nt = 0; nt < 4; ++nt) {
            const int n  = bn + wn + nt * 16 + l16;
            const float bv = bias[n];
            float rv[8];
            if (HAS_RES) {
#pragma unroll
                for (int r = 0; r < 8; ++r) {
                    const int m = bm + wm + mt * 16 + h16 * 8 + r;
                    rv[r] = res[(size_t)m * N + n];
                }
            }
#pragma unroll
            for (int r = 0; r < 8; ++r) {
                const int m = bm + wm + mt * 16 + h16 * 8 + r;
                float v = acc[mt][nt][r] + bv;
                if (HAS_RES) v += rv[r];
                if (RELU)    v = fmaxf(v, 0.0f);
                C[(size_t)m * N + n] = v;
            }
        }
}

// ---------------------------------------------------------------------------
// Flash attention: one wave per (b, h, 16-query tile). Online softmax, WMMA
// for Q*K^T and P*V.  V blocks are staged to LDS with CDNA5 async copies.
// ---------------------------------------------------------------------------
__global__ __launch_bounds__(32)
void attn_kernel(const float* __restrict__ Q, const float* __restrict__ K,
                 const float* __restrict__ V, float* __restrict__ O, int causal)
{
    const int qt = blockIdx.x * 16;
    const int h  = blockIdx.y;
    const int b  = blockIdx.z;
    const int lane = threadIdx.x;
    const int l16  = lane & 15;
    const int h16  = lane >> 4;

    __shared__ float    sV[32][DKK];    // 32-key fp32 V block (async-filled)
    __shared__ _Float16 sP[16][LDP];    // P transpose staging

    const float scale = 0.088388347648318447f;   // 1/sqrt(128)
    const size_t base = (size_t)b * SS;

    // persistent Q fragments (4 K-slices of 32 over DK=128)
    v16h qf[4];
    {
        const float* qp = Q + (base + qt + l16) * EE + h * DKK;
#pragma unroll
        for (int f = 0; f < 4; ++f) {
            const float* p = qp + f * 32 + h16 * 8;
            v16h a;
#pragma unroll
            for (int i = 0; i < 8; ++i) {
                a[i]     = (_Float16)p[i];
                a[i + 8] = (_Float16)p[i + 16];
            }
            qf[f] = a;
        }
    }

    v8f o[8];
#pragma unroll
    for (int d = 0; d < 8; ++d) o[d] = (v8f)(0.0f);
    float m[8], l[8];
#pragma unroll
    for (int r = 0; r < 8; ++r) { m[r] = -__builtin_inff(); l[r] = 0.0f; }

    const int kend = causal ? (qt + 16) : SS;
    for (int kb = 0; kb < kend; kb += 32) {
        // ---- async-stage V block (32 keys x 128 d, fp32) into LDS.
        // Guard: previous iteration's ds reads of sV must retire first
        // (async writes use a different pipe, unordered vs DS ops).
        asm volatile("s_wait_dscnt 0x0" ::: "memory");
        {
            const float* vbase = V + (base + kb) * EE + h * DKK;
#pragma unroll
            for (int c = 0; c < 32; ++c) {            // 32 chunks of 16B per lane
                const int cc  = c * 32 + lane;        // 0..1023
                const int key = cc >> 5;              // 32 chunks per key row
                const int pc  = cc & 31;
                const unsigned lds = (unsigned)(uintptr_t)&sV[key][pc * 4];
                const float* g = vbase + (size_t)key * EE + pc * 4;
                asm volatile("global_load_async_to_lds_b128 %0, %1, off"
                             :: "v"(lds), "v"(g) : "memory");
            }
        }

        // ---- scores while the copy flies: two 16x16 tiles over 32 keys
        v8f sacc[2];
#pragma unroll
        for (int st = 0; st < 2; ++st) {
            sacc[st] = (v8f)(0.0f);
            const float* kp = K + (base + kb + st * 16 + l16) * EE + h * DKK;
#pragma unroll
            for (int f = 0; f < 4; ++f) {
                const float* p = kp + f * 32 + h16 * 8;
                v16h bfr;
#pragma unroll
                for (int i = 0; i < 16; ++i) bfr[i] = (_Float16)p[i < 8 ? i : i + 8];
                sacc[st] = WMMA_F16(qf[f], bfr, sacc[st]);
            }
        }
        // ---- scale + causal mask (lane owns col kb+st*16+l16, vgpr r owns row)
#pragma unroll
        for (int r = 0; r < 8; ++r) {
            const int qrow = qt + h16 * 8 + r;
#pragma unroll
            for (int st = 0; st < 2; ++st) {
                float s = sacc[st][r] * scale;
                if (causal && (kb + st * 16 + l16) > qrow) s = -__builtin_inff();
                sacc[st][r] = s;
            }
        }
        // ---- online softmax: new row max (reduce across the 16-lane group)
        float nm[8];
#pragma unroll
        for (int r = 0; r < 8; ++r) {
            float mx = fmaxf(sacc[0][r], sacc[1][r]);
#pragma unroll
            for (int off = 1; off < 16; off <<= 1)
                mx = fmaxf(mx, __shfl_xor(mx, off, 32));
            nm[r] = fmaxf(m[r], mx);
        }
#pragma unroll
        for (int r = 0; r < 8; ++r) {
            const float corr = __expf(m[r] - nm[r]);   // exp(-inf)=0 on first block
            l[r] *= corr;
#pragma unroll
            for (int d = 0; d < 8; ++d) o[d][r] *= corr;
            m[r] = nm[r];
        }
        // ---- P = exp(s - m); partial row sums; stage P into LDS for transpose
        float psum[8];
#pragma unroll
        for (int r = 0; r < 8; ++r) psum[r] = 0.0f;
#pragma unroll
        for (int st = 0; st < 2; ++st)
#pragma unroll
            for (int r = 0; r < 8; ++r) {
                const float p = __expf(sacc[st][r] - m[r]);
                psum[r] += p;
                sP[h16 * 8 + r][st * 16 + l16] = (_Float16)p;
            }
#pragma unroll
        for (int r = 0; r < 8; ++r) {
            float s = psum[r];
#pragma unroll
            for (int off = 1; off < 16; off <<= 1)
                s += __shfl_xor(s, off, 32);
            l[r] += s;
        }
        __syncthreads();
        // ---- P as A-fragment (row = l16; K halves per A layout)
        v16h pf;
        {
            const _Float16* pp = &sP[l16][h16 * 8];
            const v8h lo = *(const v8h*)pp;
            const v8h hi = *(const v8h*)(pp + 16);
#pragma unroll
            for (int i = 0; i < 8; ++i) { pf[i] = lo[i]; pf[i + 8] = hi[i]; }
        }
        // ---- consume async V copy: O += P * V (contract 32 keys; 8 d-tiles)
        asm volatile("s_wait_asynccnt 0x0" ::: "memory");
        __syncthreads();
#pragma unroll
        for (int d = 0; d < 8; ++d) {
            v16h vf;
#pragma unroll
            for (int i = 0; i < 16; ++i)
                vf[i] = (_Float16)sV[(i < 8 ? 0 : 8) + h16 * 8 + i][d * 16 + l16];
            o[d] = WMMA_F16(pf, vf, o[d]);
        }
        __syncthreads();
    }

    // ---- normalize by row sum and write out
#pragma unroll
    for (int d = 0; d < 8; ++d) {
        const int dcol = h * DKK + d * 16 + l16;
#pragma unroll
        for (int r = 0; r < 8; ++r) {
            const int qrow = qt + h16 * 8 + r;
            O[(base + qrow) * EE + dcol] = o[d][r] / l[r];
        }
    }
}

// ---------------------------------------------------------------------------
// LayerNorm over last dim (biased variance, EPS=1e-15), one row per block.
// ---------------------------------------------------------------------------
__global__ __launch_bounds__(256)
void layernorm_kernel(const float* __restrict__ x, const float* __restrict__ g,
                      const float* __restrict__ bta, float* __restrict__ y)
{
    const int row = blockIdx.x;
    const float* xr = x + (size_t)row * EE;
    float* yr = y + (size_t)row * EE;
    __shared__ float red[8];

    float s = 0.0f;
    for (int i = threadIdx.x; i < EE; i += 256) s += xr[i];
#pragma unroll
    for (int off = 1; off < 32; off <<= 1) s += __shfl_xor(s, off, 32);
    if ((threadIdx.x & 31) == 0) red[threadIdx.x >> 5] = s;
    __syncthreads();
    float tot = 0.0f;
#pragma unroll
    for (int w = 0; w < 8; ++w) tot += red[w];
    const float mean = tot * (1.0f / EE);
    __syncthreads();

    float v = 0.0f;
    for (int i = threadIdx.x; i < EE; i += 256) {
        const float d = xr[i] - mean; v += d * d;
    }
#pragma unroll
    for (int off = 1; off < 32; off <<= 1) v += __shfl_xor(v, off, 32);
    if ((threadIdx.x & 31) == 0) red[threadIdx.x >> 5] = v;
    __syncthreads();
    float vt = 0.0f;
#pragma unroll
    for (int w = 0; w < 8; ++w) vt += red[w];
    const float rstd = rsqrtf(vt * (1.0f / EE) + 1e-15f);

    for (int i = threadIdx.x; i < EE; i += 256)
        yr[i] = g[i] * (xr[i] - mean) * rstd + bta[i];
}

// ---------------------------------------------------------------------------
// In-place softmax over V=32000 per row (online max/sum pass + normalize).
// ---------------------------------------------------------------------------
__global__ __launch_bounds__(256)
void softmax_kernel(float* __restrict__ x)
{
    const int row = blockIdx.x;
    float* xr = x + (size_t)row * VV;
    __shared__ float redm[8], reds[8];

    float m = -__builtin_inff(), l = 0.0f;
    for (int i = threadIdx.x; i < VV; i += 256) {
        const float v  = xr[i];
        const float nm = fmaxf(m, v);
        l = l * __expf(m - nm) + __expf(v - nm);
        m = nm;
    }
#pragma unroll
    for (int off = 1; off < 32; off <<= 1) {
        const float om = __shfl_xor(m, off, 32);
        const float ol = __shfl_xor(l, off, 32);
        const float nm = fmaxf(m, om);
        l = l * __expf(m - nm) + ol * __expf(om - nm);
        m = nm;
    }
    if ((threadIdx.x & 31) == 0) { redm[threadIdx.x >> 5] = m; reds[threadIdx.x >> 5] = l; }
    __syncthreads();
    float fm = -__builtin_inff(), fl = 0.0f;
#pragma unroll
    for (int w = 0; w < 8; ++w) {
        const float om = redm[w], ol = reds[w];
        const float nm = fmaxf(fm, om);
        fl = fl * __expf(fm - nm) + ol * __expf(om - nm);
        fm = nm;
    }
    const float inv = 1.0f / fl;
    for (int i = threadIdx.x; i < VV; i += 256)
        xr[i] = __expf(xr[i] - fm) * inv;
}

// ---------------------------------------------------------------------------
// Host orchestration
// ---------------------------------------------------------------------------
enum GemmMode { GEMM_PLAIN = 0, GEMM_RELU = 1, GEMM_RES = 2 };

static inline void launch_gemm(const float* A, const float* B, const float* bias,
                               const float* res, float* C, int M, int N, int K,
                               GemmMode mode, hipStream_t s)
{
    dim3 g(N / BN, M / BM), blk(256);
    switch (mode) {
    case GEMM_PLAIN:
        gemm_wmma_kernel<false, false><<<g, blk, 0, s>>>(A, B, bias, res, C, M, N, K);
        break;
    case GEMM_RELU:
        gemm_wmma_kernel<false, true ><<<g, blk, 0, s>>>(A, B, bias, res, C, M, N, K);
        break;
    case GEMM_RES:
        gemm_wmma_kernel<true,  false><<<g, blk, 0, s>>>(A, B, bias, res, C, M, N, K);
        break;
    }
}

extern "C" void kernel_launch(void* const* d_in, const int* in_sizes, int n_in,
                              void* d_out, int out_size, void* d_ws, size_t ws_size,
                              hipStream_t stream)
{
    const int*   src     = (const int*)  d_in[0];
    const int*   trg     = (const int*)  d_in[1];
    const float* emb_enc = (const float*)d_in[2];
    const float* emb_dec = (const float*)d_in[3];
    const float* pos     = (const float*)d_in[4];
    const float* mha_w   = (const float*)d_in[5];
    const float* mha_b   = (const float*)d_in[6];
    const float* ffn_w1  = (const float*)d_in[7];
    const float* ffn_b1  = (const float*)d_in[8];
    const float* ffn_w2  = (const float*)d_in[9];
    const float* ffn_b2  = (const float*)d_in[10];
    const float* norm_g  = (const float*)d_in[11];
    const float* norm_b  = (const float*)d_in[12];
    const float* wout_w  = (const float*)d_in[13];
    const float* wout_b  = (const float*)d_in[14];
    float* out = (float*)d_out;

    auto W  = [&](int blk, int j) { return mha_w + ((size_t)(blk * 4 + j)) * EE * EE; };
    auto Bv = [&](int blk, int j) { return mha_b + ((size_t)(blk * 4 + j)) * EE; };

    // workspace layout (fp32): 9 buffers of B*S*E (hid uses 2 of them)
    const size_t A1 = (size_t)MM * EE;   // 2,097,152 floats
    float* ws    = (float*)d_ws;
    float* enc_x = ws;
    float* dec_x = ws + 1 * A1;
    float* qb    = ws + 2 * A1;
    float* kb    = ws + 3 * A1;
    float* vb    = ws + 4 * A1;
    float* attn  = ws + 5 * A1;
    float* tmp   = ws + 6 * A1;
    float* hid   = ws + 7 * A1;          // size 2*A1 = M*HID

    const dim3 rowsG(MM), b256(256);
    const dim3 attnG(SS / 16, HH, BB), b32(32);

    // ---- embeddings
    embed_kernel<<<rowsG, b256, 0, stream>>>(src, emb_enc, pos, enc_x);
    embed_kernel<<<rowsG, b256, 0, stream>>>(trg, emb_dec, pos, dec_x);

    // ---- encoder self-attention (mha block 0)
    launch_gemm(enc_x, W(0,0), Bv(0,0), nullptr, qb, MM, EE, EE, GEMM_PLAIN, stream);
    launch_gemm(enc_x, W(0,1), Bv(0,1), nullptr, kb, MM, EE, EE, GEMM_PLAIN, stream);
    launch_gemm(enc_x, W(0,2), Bv(0,2), nullptr, vb, MM, EE, EE, GEMM_PLAIN, stream);
    attn_kernel<<<attnG, b32, 0, stream>>>(qb, kb, vb, attn, 0);
    launch_gemm(attn, W(0,3), Bv(0,3), enc_x, tmp, MM, EE, EE, GEMM_RES, stream);
    layernorm_kernel<<<rowsG, b256, 0, stream>>>(tmp, norm_g + 0*EE, norm_b + 0*EE, enc_x);

    // ---- encoder FFN (block 0)
    launch_gemm(enc_x, ffn_w1, ffn_b1, nullptr, hid, MM, HIDD, EE, GEMM_RELU, stream);
    launch_gemm(hid, ffn_w2, ffn_b2, enc_x, tmp, MM, EE, HIDD, GEMM_RES, stream);
    layernorm_kernel<<<rowsG, b256, 0, stream>>>(tmp, norm_g + 1*EE, norm_b + 1*EE, enc_x);

    // ---- decoder masked self-attention (mha block 1)
    launch_gemm(dec_x, W(1,0), Bv(1,0), nullptr, qb, MM, EE, EE, GEMM_PLAIN, stream);
    launch_gemm(dec_x, W(1,1), Bv(1,1), nullptr, kb, MM, EE, EE, GEMM_PLAIN, stream);
    launch_gemm(dec_x, W(1,2), Bv(1,2), nullptr, vb, MM, EE, EE, GEMM_PLAIN, stream);
    attn_kernel<<<attnG, b32, 0, stream>>>(qb, kb, vb, attn, 1);
    launch_gemm(attn, W(1,3), Bv(1,3), dec_x, tmp, MM, EE, EE, GEMM_RES, stream);
    layernorm_kernel<<<rowsG, b256, 0, stream>>>(tmp, norm_g + 2*EE, norm_b + 2*EE, dec_x);

    // ---- decoder cross-attention (mha block 2; K,V from encoder output)
    launch_gemm(dec_x, W(2,0), Bv(2,0), nullptr, qb, MM, EE, EE, GEMM_PLAIN, stream);
    launch_gemm(enc_x, W(2,1), Bv(2,1), nullptr, kb, MM, EE, EE, GEMM_PLAIN, stream);
    launch_gemm(enc_x, W(2,2), Bv(2,2), nullptr, vb, MM, EE, EE, GEMM_PLAIN, stream);
    attn_kernel<<<attnG, b32, 0, stream>>>(qb, kb, vb, attn, 0);
    launch_gemm(attn, W(2,3), Bv(2,3), dec_x, tmp, MM, EE, EE, GEMM_RES, stream);
    layernorm_kernel<<<rowsG, b256, 0, stream>>>(tmp, norm_g + 3*EE, norm_b + 3*EE, dec_x);

    // ---- decoder FFN (block 1)
    launch_gemm(dec_x, ffn_w1 + (size_t)EE*HIDD, ffn_b1 + HIDD, nullptr, hid,
                MM, HIDD, EE, GEMM_RELU, stream);
    launch_gemm(hid, ffn_w2 + (size_t)HIDD*EE, ffn_b2 + EE, dec_x, tmp,
                MM, EE, HIDD, GEMM_RES, stream);
    layernorm_kernel<<<rowsG, b256, 0, stream>>>(tmp, norm_g + 4*EE, norm_b + 4*EE, dec_x);

    // ---- output projection + softmax (in-place in d_out)
    launch_gemm(dec_x, wout_w, wout_b, nullptr, out, MM, VV, EE, GEMM_PLAIN, stream);
    softmax_kernel<<<rowsG, b256, 0, stream>>>(out);
}
